// BalancedHamiltonLayer_19731079758486
// MI455X (gfx1250) — compile-verified, matrix operand order checked by reading, not run on hardware
//
#include <hip/hip_runtime.h>
#include <stdint.h>

typedef _Float16 f16;
typedef __attribute__((ext_vector_type(4)))  _Float16 v4h;
typedef __attribute__((ext_vector_type(8)))  _Float16 v8h;
typedef __attribute__((ext_vector_type(16))) _Float16 v16h;
typedef __attribute__((ext_vector_type(8)))  float    v8f;
typedef __attribute__((ext_vector_type(4)))  int      v4i;

#define DIM_K 2048
#define DIM_N 2048
#define M_TOTAL (4 * 4096)   // B*T = 16384
#define RANK 8

#define BM 128
#define BN 128
#define BK 32
#define LDS_STRIDE 40        // halves per row: 80B rows -> every row 16B aligned
#define NTHREADS 256

#if defined(__has_builtin)
#if __has_builtin(__builtin_amdgcn_global_load_async_to_lds_b128)
#define HAVE_ASYNC_LDS 1
#else
#define HAVE_ASYNC_LDS 0
#endif
#else
#define HAVE_ASYNC_LDS 0
#endif

__device__ __forceinline__ void async_copy_b128(void* lds_dst, const void* gsrc) {
#if HAVE_ASYNC_LDS
  // ASYNCcnt-tracked DMA: global -> LDS without a VGPR round trip.
  __builtin_amdgcn_global_load_async_to_lds_b128((v4i*)gsrc, (v4i*)lds_dst,
                                                 /*offset=*/0, /*cpol=*/0);
#else
  *(uint4*)lds_dst = *(const uint4*)gsrc;
#endif
}

__device__ __forceinline__ void wait_async_zero() {
#if HAVE_ASYNC_LDS
#if __has_builtin(__builtin_amdgcn_s_wait_asynccnt)
  __builtin_amdgcn_s_wait_asynccnt(0);
#else
  asm volatile("s_wait_asynccnt 0" ::: "memory");
#endif
#endif
}

// ---------------------------------------------------------------------------
// Kernel 1: Wt_hi/Wt_lo[e][d] = f16 hi/lo split of (sum_r factors[r][d][e])/sqrt(8)
// Tiled 64x64 transpose through LDS: coalesced reads along e, writes along d.
// ---------------------------------------------------------------------------
__global__ __launch_bounds__(NTHREADS) void wsum_split_transpose_kernel(
    const float* __restrict__ factors, f16* __restrict__ wth,
    f16* __restrict__ wtl) {
  __shared__ __align__(16) float S[64 * 68];   // [d_local][e_local], padded
  const int tid = threadIdx.x;
  const int d0 = blockIdx.x * 64;
  const int e0 = blockIdx.y * 64;

#pragma unroll
  for (int it = 0; it < 4; ++it) {             // 64 rows x 16 float4 = 1024
    const int v  = tid + it * NTHREADS;
    const int i  = v >> 4;                     // local d row
    const int c4 = v & 15;                     // float4 along e
    const float* p = factors + (size_t)(d0 + i) * DIM_N + e0 + c4 * 4;
    float4 a = *(const float4*)p;
#pragma unroll
    for (int r = 1; r < RANK; ++r) {
      float4 t = *(const float4*)(p + (size_t)r * DIM_K * DIM_N);
      a.x += t.x; a.y += t.y; a.z += t.z; a.w += t.w;
    }
    const float s = 0.35355339059327373f;      // 1/sqrt(8)
    a.x *= s; a.y *= s; a.z *= s; a.w *= s;
    *(float4*)&S[i * 68 + c4 * 4] = a;
  }
  __syncthreads();

#pragma unroll
  for (int it = 0; it < 16; ++it) {            // 4096 elements
    const int v  = tid + it * NTHREADS;
    const int er = v >> 6;                     // local e row (output row)
    const int dc = v & 63;                     // local d col (contiguous writes)
    const float val = S[dc * 68 + er];
    const f16 hi = (f16)val;
    const f16 lo = (f16)(val - (float)hi);
    const size_t o = (size_t)(e0 + er) * DIM_K + (d0 + dc);
    wth[o] = hi;
    wtl[o] = lo;
  }
}

// ---------------------------------------------------------------------------
// Kernel 2: out = x @ Wsum + bias via split-f16 WMMA (16x16x32, f32 accum):
//   x*W ~= x_hi*w_hi + x_lo*w_hi + x_hi*w_lo   (error ~2^-22, fp32-class)
// Block: 256 thr = 8 waves (2Mx4N), 128x128 tile, BK=32 stage.
// Wave: 64x32 region = 4x2 tiles of 16x16 -> 24 WMMAs per K-stage.
// ---------------------------------------------------------------------------
__global__ __launch_bounds__(NTHREADS) void gemm_wmma_f16split(
    const float* __restrict__ X, const f16* __restrict__ Wth,
    const f16* __restrict__ Wtl, const float* __restrict__ bias,
    float* __restrict__ C) {
  __shared__ __align__(16) f16 Ah[BM * LDS_STRIDE];  // x hi, [m][k]
  __shared__ __align__(16) f16 Al[BM * LDS_STRIDE];  // x lo
  __shared__ __align__(16) f16 Bh[BN * LDS_STRIDE];  // W^T hi, [n][k]
  __shared__ __align__(16) f16 Bl[BN * LDS_STRIDE];  // W^T lo

  const int tid  = threadIdx.x;
  const int lane = tid & 31;
  const int wv   = tid >> 5;
  const int wm0  = (wv >> 2) * 64;   // wave M origin (0,64)
  const int wn0  = (wv & 3) * 32;    // wave N origin (0,32,64,96)
  const int lr   = lane & 15;
  const int lh   = lane >> 4;        // half-wave select

  const int m0 = blockIdx.y * BM;
  const int n0 = blockIdx.x * BN;

  v8f acc[4][2];
#pragma unroll
  for (int mt = 0; mt < 4; ++mt)
#pragma unroll
    for (int nt = 0; nt < 2; ++nt) acc[mt][nt] = (v8f)0.0f;

  for (int k0 = 0; k0 < DIM_K; k0 += BK) {
    // --- B tiles: pure contiguous f16 copy -> async global->LDS DMA ---
    // 128 rows x 64B (hi) + same (lo); 4x b128 per row; 2+2 chunks/thread
#pragma unroll
    for (int it = 0; it < 2; ++it) {
      const int v   = tid + it * NTHREADS;     // 0..511
      const int row = v >> 2;
      const int c   = v & 3;
      const size_t goff = (size_t)(n0 + row) * DIM_K + k0 + c * 8;
      const int loff = row * LDS_STRIDE + c * 8;
      async_copy_b128(&Bh[loff], Wth + goff);
      async_copy_b128(&Bl[loff], Wtl + goff);
    }
    // --- A tile: load fp32 x, split hi/lo f16 in-register, store to LDS ---
#pragma unroll
    for (int it = 0; it < 4; ++it) {           // 128 rows x 8 float4 = 1024
      const int v   = tid + it * NTHREADS;
      const int row = v >> 3;
      const int c4  = v & 7;
      float4 d = *(const float4*)(X + (size_t)(m0 + row) * DIM_K + k0 + c4 * 4);
      v4h hi, lo;
      hi.x = (f16)d.x; lo.x = (f16)(d.x - (float)hi.x);
      hi.y = (f16)d.y; lo.y = (f16)(d.y - (float)hi.y);
      hi.z = (f16)d.z; lo.z = (f16)(d.z - (float)hi.z);
      hi.w = (f16)d.w; lo.w = (f16)(d.w - (float)hi.w);
      const int loff = row * LDS_STRIDE + c4 * 4;
      *(v4h*)&Ah[loff] = hi;
      *(v4h*)&Al[loff] = lo;
    }
    wait_async_zero();
    __syncthreads();

    // --- fragments: 16-bit A 16x32 layout = two 8-half runs per lane:
    //     k in [8*lh, 8*lh+8) and [16+8*lh, 16+8*lh+8); B mirrored over W^T.
    v16h afh[4], afl[4];
#pragma unroll
    for (int mt = 0; mt < 4; ++mt) {
      const int m = wm0 + mt * 16 + lr;
      v8h h0 = *(const v8h*)&Ah[m * LDS_STRIDE + 8 * lh];
      v8h h1 = *(const v8h*)&Ah[m * LDS_STRIDE + 16 + 8 * lh];
      afh[mt] = __builtin_shufflevector(h0, h1, 0, 1, 2, 3, 4, 5, 6, 7, 8, 9,
                                        10, 11, 12, 13, 14, 15);
      v8h l0 = *(const v8h*)&Al[m * LDS_STRIDE + 8 * lh];
      v8h l1 = *(const v8h*)&Al[m * LDS_STRIDE + 16 + 8 * lh];
      afl[mt] = __builtin_shufflevector(l0, l1, 0, 1, 2, 3, 4, 5, 6, 7, 8, 9,
                                        10, 11, 12, 13, 14, 15);
    }
    v16h bfh[2], bfl[2];
#pragma unroll
    for (int nt = 0; nt < 2; ++nt) {
      const int n = wn0 + nt * 16 + lr;
      v8h h0 = *(const v8h*)&Bh[n * LDS_STRIDE + 8 * lh];
      v8h h1 = *(const v8h*)&Bh[n * LDS_STRIDE + 16 + 8 * lh];
      bfh[nt] = __builtin_shufflevector(h0, h1, 0, 1, 2, 3, 4, 5, 6, 7, 8, 9,
                                        10, 11, 12, 13, 14, 15);
      v8h l0 = *(const v8h*)&Bl[n * LDS_STRIDE + 8 * lh];
      v8h l1 = *(const v8h*)&Bl[n * LDS_STRIDE + 16 + 8 * lh];
      bfl[nt] = __builtin_shufflevector(l0, l1, 0, 1, 2, 3, 4, 5, 6, 7, 8, 9,
                                        10, 11, 12, 13, 14, 15);
    }

#pragma unroll
    for (int mt = 0; mt < 4; ++mt)
#pragma unroll
      for (int nt = 0; nt < 2; ++nt) {
        acc[mt][nt] = __builtin_amdgcn_wmma_f32_16x16x32_f16(
            false, afh[mt], false, bfh[nt], (short)0, acc[mt][nt], false, false);
        acc[mt][nt] = __builtin_amdgcn_wmma_f32_16x16x32_f16(
            false, afl[mt], false, bfh[nt], (short)0, acc[mt][nt], false, false);
        acc[mt][nt] = __builtin_amdgcn_wmma_f32_16x16x32_f16(
            false, afh[mt], false, bfl[nt], (short)0, acc[mt][nt], false, false);
      }
    __syncthreads();
  }

  // --- epilogue: C/D layout: lanes 0-15 hold M=v, lanes 16-31 hold M=v+8 ---
#pragma unroll
  for (int nt = 0; nt < 2; ++nt) {
    const int n = n0 + wn0 + nt * 16 + lr;
    const float bv = bias[n];
#pragma unroll
    for (int mt = 0; mt < 4; ++mt) {
      const int mbase = m0 + wm0 + mt * 16 + lh * 8;
#pragma unroll
      for (int v = 0; v < 8; ++v)
        C[(size_t)(mbase + v) * DIM_N + n] = acc[mt][nt][v] + bv;
    }
  }
}

// ---------------------------------------------------------------------------
extern "C" void kernel_launch(void* const* d_in, const int* in_sizes, int n_in,
                              void* d_out, int out_size, void* d_ws, size_t ws_size,
                              hipStream_t stream) {
  (void)in_sizes; (void)n_in; (void)out_size; (void)ws_size;
  const float* x       = (const float*)d_in[0];   // [4,4096,2048] fp32
  const float* factors = (const float*)d_in[1];   // [8,2048,2048] fp32
  const float* bias    = (const float*)d_in[2];   // [2048] fp32
  float* out = (float*)d_out;                     // [4,4096,2048] fp32

  f16* wth = (f16*)d_ws;                          // W^T hi, 8 MB
  f16* wtl = wth + (size_t)DIM_K * DIM_N;         // W^T lo, 8 MB

  // 1) fold rank-sum + scale, split to f16 hi/lo, transpose (once, 16 MB)
  wsum_split_transpose_kernel<<<dim3(DIM_K / 64, DIM_N / 64), 256, 0, stream>>>(
      factors, wth, wtl);

  // 2) split-f16 WMMA GEMM: out = x @ Wsum + bias
  dim3 grid(DIM_N / BN, M_TOTAL / BM);
  gemm_wmma_f16split<<<grid, 256, 0, stream>>>(x, wth, wtl, bias, out);
}
